// GraphBased_84456236908813
// MI455X (gfx1250) — matrix operand order repaired and checked by reading, not compiled, not run on hardware
//
#include <hip/hip_runtime.h>
#include <hip/hip_bf16.h>
#include <math.h>

typedef __attribute__((ext_vector_type(16))) _Float16 v16h;
typedef __attribute__((ext_vector_type(8)))  _Float16 v8h;
typedef __attribute__((ext_vector_type(8)))  float    v8f;

#define N_NODES 4096
#define NT      256      // 16x16 tiles per matrix dimension (4096/16)
#define FEAT    50
#define FPAD    64       // K padded to 64 for two 16x16x32 f16 WMMAs
#define NCAND   256

__device__ __forceinline__ float leaky(float x){ return x >= 0.0f ? x : 0.01f*x; }

// ---------------- kernel 0: zero the distance histogram ----------------
__global__ void k_init(int* __restrict__ hist){
  for (int i = threadIdx.x; i < NCAND + 2; i += blockDim.x) hist[i] = 0;
}

// ---------------- kernel 1: conv1(1->20,5x5) + relu + maxpool2 ----------------
// x [4096][28][28] -> p1 [4096][20][12][12]
__global__ void k_conv1(const float* __restrict__ x, const float* __restrict__ w,
                        const float* __restrict__ b, float* __restrict__ p1){
  int t = blockIdx.x * blockDim.x + threadIdx.x;
  const int total = N_NODES * 20 * 144;
  if (t >= total) return;
  int n  = t / (20*144);
  int r  = t % (20*144);
  int co = r / 144;
  int p  = r % 144;
  int oy = p / 12, ox = p % 12;
  const float* img = x + (size_t)n * 784;
  int y0 = oy*2, x0 = ox*2;
  float win[6][6];
  #pragma unroll
  for (int wy = 0; wy < 6; ++wy)
    #pragma unroll
    for (int wx = 0; wx < 6; ++wx)
      win[wy][wx] = img[(y0+wy)*28 + (x0+wx)];
  float a00=0.f, a01=0.f, a10=0.f, a11=0.f;
  const float* wc = w + co*25;
  #pragma unroll
  for (int ky = 0; ky < 5; ++ky)
    #pragma unroll
    for (int kx = 0; kx < 5; ++kx){
      float wv = wc[ky*5+kx];
      a00 = fmaf(wv, win[ky  ][kx  ], a00);
      a01 = fmaf(wv, win[ky  ][kx+1], a01);
      a10 = fmaf(wv, win[ky+1][kx  ], a10);
      a11 = fmaf(wv, win[ky+1][kx+1], a11);
    }
  float bb = b[co];
  a00 = fmaxf(a00+bb, 0.f); a01 = fmaxf(a01+bb, 0.f);
  a10 = fmaxf(a10+bb, 0.f); a11 = fmaxf(a11+bb, 0.f);
  p1[t] = fmaxf(fmaxf(a00,a01), fmaxf(a10,a11));
}

// ---------------- kernel 2: conv2(20->50,5x5) + relu + maxpool2 ----------------
// p1 [n][20][12][12] -> p2 [n][800]  (800 = 50ch * 4 * 4, matching reshape order)
__global__ void k_conv2(const float* __restrict__ p1, const float* __restrict__ w,
                        const float* __restrict__ b, float* __restrict__ p2){
  __shared__ float img[20*144];
  int n = blockIdx.x;
  for (int i = threadIdx.x; i < 20*144; i += blockDim.x)
    img[i] = p1[(size_t)n*20*144 + i];
  __syncthreads();
  for (int o = threadIdx.x; o < 800; o += blockDim.x){
    int co = o / 16, p = o % 16;
    int oy = p / 4, ox = p % 4;
    int y0 = oy*2, x0 = ox*2;
    float a00=0.f, a01=0.f, a10=0.f, a11=0.f;
    for (int ci = 0; ci < 20; ++ci){
      float win[6][6];
      const float* im = img + ci*144;
      #pragma unroll
      for (int wy = 0; wy < 6; ++wy)
        #pragma unroll
        for (int wx = 0; wx < 6; ++wx)
          win[wy][wx] = im[(y0+wy)*12 + (x0+wx)];
      const float* wc = w + (co*20 + ci)*25;
      #pragma unroll
      for (int ky = 0; ky < 5; ++ky)
        #pragma unroll
        for (int kx = 0; kx < 5; ++kx){
          float wv = wc[ky*5+kx];
          a00 = fmaf(wv, win[ky  ][kx  ], a00);
          a01 = fmaf(wv, win[ky  ][kx+1], a01);
          a10 = fmaf(wv, win[ky+1][kx  ], a10);
          a11 = fmaf(wv, win[ky+1][kx+1], a11);
        }
    }
    float bb = b[co];
    a00 = fmaxf(a00+bb,0.f); a01 = fmaxf(a01+bb,0.f);
    a10 = fmaxf(a10+bb,0.f); a11 = fmaxf(a11+bb,0.f);
    p2[(size_t)n*800 + o] = fmaxf(fmaxf(a00,a01), fmaxf(a10,a11));
  }
}

// ---------------- kernel 3: FC 800->50 + relu; emit f16 padded copy + sumsq ----------------
__global__ void k_fc(const float* __restrict__ p2, const float* __restrict__ w,
                     const float* __restrict__ b, float* __restrict__ h,
                     _Float16* __restrict__ hpad, float* __restrict__ sq){
  __shared__ float row[800];
  __shared__ float part[64];
  int n = blockIdx.x, t = threadIdx.x;           // blockDim = 64
  for (int i = t; i < 800; i += 64) row[i] = p2[(size_t)n*800 + i];
  __syncthreads();
  float hv = 0.f;
  if (t < FEAT){
    float acc = b[t];
    const float* wr = w + (size_t)t*800;
    for (int k = 0; k < 800; ++k) acc = fmaf(row[k], wr[k], acc);
    hv = fmaxf(acc, 0.f);
    h[(size_t)n*FEAT + t] = hv;
  }
  _Float16 hh = (t < FEAT) ? (_Float16)hv : (_Float16)0.f;
  hpad[(size_t)n*FPAD + t] = hh;                 // zero pad cols 50..63
  float hf = (float)hh;                          // sq from f16-rounded values
  part[t] = hf*hf;
  __syncthreads();
  #pragma unroll
  for (int s = 32; s > 0; s >>= 1){
    if (t < s) part[t] += part[t+s];
    __syncthreads();
  }
  if (t == 0) sq[n] = part[0];
}

// ---------------- shared WMMA Gram-tile helper (deterministic, f16 -> f32 acc) ----------------
// C[r] holds G[ti*16 + r + 8*(lane/16)][tj*16 + lane%16]
__device__ __forceinline__ v8f gram_tile(const _Float16* __restrict__ hp,
                                         int ti, int tj, int lane){
  int m = lane & 15;
  int half = lane >> 4;
  v8f c = {};
  const _Float16* rowa = hp + (size_t)(ti*16 + m) * FPAD;
  const _Float16* rowb = hp + (size_t)(tj*16 + m) * FPAD;
  #pragma unroll
  for (int k0 = 0; k0 < FPAD; k0 += 32){
    // A 16x32 f16: lane holds K = k0 + {half*8 + e, 16 + half*8 + e}
    v8h alo = *(const v8h*)(rowa + k0 + half*8);
    v8h ahi = *(const v8h*)(rowa + k0 + half*8 + 16);
    // B 32x16 f16 (= h^T tile): lane n = m, K = k0 + {half*8 + e, 16 + half*8 + e}
    v8h blo = *(const v8h*)(rowb + k0 + half*8);
    v8h bhi = *(const v8h*)(rowb + k0 + half*8 + 16);
    v16h a, bm;
    #pragma unroll
    for (int e = 0; e < 8; ++e){
      a[e] = alo[e]; a[e+8] = ahi[e];
      bm[e] = blo[e]; bm[e+8] = bhi[e];
    }
    c = __builtin_amdgcn_wmma_f32_16x16x32_f16(false, a, false, bm,
                                               (short)0, c, false, false);
  }
  return c;
}

// ---------------- kernel 4: Gram via WMMA + fused distance histogram ----------------
// replaces the reference's 16.7M-element sort: counts[k] is just a 257-bin histogram
__global__ void k_gram_hist(const _Float16* __restrict__ hpad,
                            const float* __restrict__ sq,
                            int* __restrict__ hist){
  __shared__ int lh[NCAND + 2];
  for (int i = threadIdx.x; i < NCAND + 2; i += blockDim.x) lh[i] = 0;
  __syncthreads();
  int wave = threadIdx.x >> 5;
  int lane = threadIdx.x & 31;
  int tile = blockIdx.x * 8 + wave;              // grid = NT*NT/8 blocks, 8 waves each
  int ti = tile / NT, tj = tile % NT;
  v8f c = gram_tile(hpad, ti, tj, lane);
  int m = lane & 15, half = lane >> 4;
  int j = tj*16 + m;
  float sqj = sq[j];
  #pragma unroll
  for (int r = 0; r < 8; ++r){
    int i = ti*16 + r + 8*half;
    if (i == j) continue;                        // diagonal -> inf in reference
    float d2 = sq[i] + sqj - 2.0f*c[r];
    float d  = sqrtf(fmaxf(d2, 0.0f));
    // smallest k with d < 3.5 + 0.5k (exact-compare fixups match searchsorted 'left')
    int k0 = (int)floorf((d - 3.5f)*2.0f) + 1;
    if (k0 < 0) k0 = 0;
    while (k0 > 0 && d < 3.5f + 0.5f*(float)(k0-1)) --k0;
    while (k0 <= NCAND && d >= 3.5f + 0.5f*(float)k0) ++k0;
    if (k0 <= NCAND) atomicAdd(&lh[k0], 1);
  }
  __syncthreads();
  for (int i = threadIdx.x; i < NCAND + 2; i += blockDim.x)
    if (lh[i]) atomicAdd(&hist[i], lh[i]);
}

// ---------------- kernel 5: prefix sum + radius selection ----------------
__global__ void k_select(const int* __restrict__ hist, float* __restrict__ sel){
  int cum = 0, kstar = -1, cnt = 0;
  const float need = 0.1f * (float)N_NODES;      // matches counts >= 0.1*n in f32
  for (int k = 0; k < NCAND; ++k){
    cum += hist[k];
    if (kstar < 0 && (float)cum >= need){ kstar = k; cnt = cum; }
  }
  if (kstar < 0){ kstar = 0; cnt = hist[0]; }    // argmax(all false) == 0
  sel[0] = 3.5f + 0.5f*(float)kstar;             // thr
  sel[1] = (float)cnt;                           // nnz(A)  (exact in f32, < 2^24)
}

// ---------------- kernel 6: masked SpMM  agg = A @ h, deg = rowsum(A) ----------------
// Recomputes Gram tiles with WMMA; A is very sparse (>=410 edges), so edge
// accumulation into per-wave LDS accumulators is cheap; combined in fixed order.
__global__ void k_spmm(const _Float16* __restrict__ hpad,
                       const float* __restrict__ sq,
                       const float* __restrict__ h,
                       const float* __restrict__ sel,
                       float* __restrict__ agg,
                       float* __restrict__ deg){
  __shared__ float acc[8][16][FEAT];
  __shared__ unsigned int dg[8][16];
  for (int i = threadIdx.x; i < 8*16*FEAT; i += blockDim.x) ((float*)acc)[i] = 0.f;
  for (int i = threadIdx.x; i < 8*16;      i += blockDim.x) ((unsigned int*)dg)[i] = 0u;
  __syncthreads();
  int wave = threadIdx.x >> 5, lane = threadIdx.x & 31;
  int m = lane & 15, half = lane >> 4;
  int ti = blockIdx.x;
  float thr = sel[0];
  for (int tj = wave; tj < NT; tj += 8){         // uniform per wave: EXEC all-1 at WMMA
    v8f c = gram_tile(hpad, ti, tj, lane);
    int j = tj*16 + m;
    float sqj = sq[j];
    #pragma unroll
    for (int r = 0; r < 8; ++r){
      int il = r + 8*half;
      int i  = ti*16 + il;
      if (i == j) continue;
      float d2 = sq[i] + sqj - 2.0f*c[r];
      float d  = sqrtf(fmaxf(d2, 0.0f));
      if (d < thr){                              // identical predicate to histogram pass
        atomicAdd(&dg[wave][il], 1u);
        const float* hj = h + (size_t)j*FEAT;
        for (int cc = 0; cc < FEAT; ++cc)
          atomicAdd(&acc[wave][il][cc], hj[cc]); // ds_add_f32
      }
    }
  }
  __syncthreads();
  for (int idx = threadIdx.x; idx < 16*FEAT; idx += blockDim.x){
    int il = idx / FEAT, cc = idx % FEAT;
    float s = 0.f;
    #pragma unroll
    for (int w = 0; w < 8; ++w) s += acc[w][il][cc];
    agg[(size_t)(ti*16 + il)*FEAT + cc] = s;
  }
  if (threadIdx.x < 16){
    unsigned int s = 0;
    #pragma unroll
    for (int w = 0; w < 8; ++w) s += dg[w][threadIdx.x];
    deg[ti*16 + threadIdx.x] = (float)s;
  }
}

// ---------------- kernel 7: SAGE embed  Z = leaky((agg/deg)Wrel^T + brel + h Wroot^T) ----------------
__global__ void k_sage(const float* __restrict__ h, const float* __restrict__ agg,
                       const float* __restrict__ deg,
                       const float* __restrict__ wrel, const float* __restrict__ brel,
                       const float* __restrict__ wroot,
                       float* __restrict__ Z){
  __shared__ float hr[FEAT], mr[FEAT];
  int n = blockIdx.x, t = threadIdx.x;           // blockDim = 64
  float dgc = fmaxf(deg[n], 1.0f);
  if (t < FEAT){
    hr[t] = h[(size_t)n*FEAT + t];
    mr[t] = agg[(size_t)n*FEAT + t] / dgc;
  }
  __syncthreads();
  if (t < FEAT){
    float a = brel[t];
    const float* wr = wrel  + t*FEAT;
    const float* wo = wroot + t*FEAT;
    for (int k = 0; k < FEAT; ++k) a += mr[k]*wr[k] + hr[k]*wo[k];
    Z[(size_t)n*FEAT + t] = leaky(a);
  }
}

// ---------------- kernel 8: diff-pool (s==ones) + head MLP + softmax ----------------
// softmax over the length-1 axis of S makes s identically 1: Xp = colsum(Z),
// adjp = nnz(A), l1 = sqrt(N^2 - nnz)/N^2. The pool/mlp branch is dead code.
__global__ void k_final(const float* __restrict__ Z, const float* __restrict__ sel,
                        const float* __restrict__ wrel, const float* __restrict__ brel,
                        const float* __restrict__ wroot,
                        const float* __restrict__ l1w, const float* __restrict__ l1b,
                        const float* __restrict__ l2w, const float* __restrict__ l2b,
                        float* __restrict__ out, int out_size){
  __shared__ float xp[FEAT], x2[FEAT], v1[25];
  int t = threadIdx.x;                           // blockDim = 64
  if (t < FEAT){
    float s = 0.f;
    for (int i = 0; i < N_NODES; ++i) s += Z[(size_t)i*FEAT + t];  // fixed order
    xp[t] = s;
  }
  __syncthreads();
  float adjp = sel[1];
  float degp = fmaxf(adjp, 1.0f);
  if (t < FEAT){
    float a = brel[t];
    const float* wr = wrel  + t*FEAT;
    const float* wo = wroot + t*FEAT;
    for (int k = 0; k < FEAT; ++k){
      float mk = (adjp * xp[k]) / degp;
      a += mk*wr[k] + xp[k]*wo[k];
    }
    x2[t] = leaky(a);
  }
  __syncthreads();
  if (t < 25){
    float a = l1b[t];
    for (int k = 0; k < FEAT; ++k) a += x2[k]*l1w[t*FEAT + k];
    v1[t] = leaky(a);
  }
  __syncthreads();
  if (t == 0){
    float v2[2];
    #pragma unroll
    for (int jj = 0; jj < 2; ++jj){
      float a = l2b[jj];
      for (int k = 0; k < 25; ++k) a += v1[k]*l2w[jj*25 + k];
      v2[jj] = leaky(a);
    }
    float mx = fmaxf(v2[0], v2[1]);
    float e0 = expf(v2[0]-mx), e1 = expf(v2[1]-mx);
    float se = e0 + e1;
    float p0 = e0/se, p1 = e1/se;
    float nn = (float)N_NODES * (float)N_NODES;  // 2^24, exact
    float l1 = sqrtf(nn - adjp) / nn;
    if (out_size > 0) out[0] = fmaxf(p0, p1);
    if (out_size > 1) out[1] = (p1 > p0) ? 1.0f : 0.0f;  // argmax, ties -> 0
    if (out_size > 2) out[2] = l1;
  }
}

extern "C" void kernel_launch(void* const* d_in, const int* in_sizes, int n_in,
                              void* d_out, int out_size, void* d_ws, size_t ws_size,
                              hipStream_t stream){
  const float* x       = (const float*)d_in[0];
  const float* c1w     = (const float*)d_in[1];
  const float* c1b     = (const float*)d_in[2];
  const float* c2w     = (const float*)d_in[3];
  const float* c2b     = (const float*)d_in[4];
  const float* fcw     = (const float*)d_in[5];
  const float* fcb     = (const float*)d_in[6];
  const float* erel_w  = (const float*)d_in[7];
  const float* erel_b  = (const float*)d_in[8];
  const float* eroot_w = (const float*)d_in[9];
  // d_in[10..14] (pool_rel_w/b, pool_root_w, mlp_w/b) are mathematically dead:
  // softmax over the singleton last axis of S is identically 1.
  const float* l1w     = (const float*)d_in[15];
  const float* l1b     = (const float*)d_in[16];
  const float* l2w     = (const float*)d_in[17];
  const float* l2b     = (const float*)d_in[18];
  (void)in_sizes; (void)n_in; (void)ws_size;

  char* ws = (char*)d_ws;
  size_t off = 0;
  auto alloc = [&](size_t bytes)->void*{
    void* p = ws + off;
    off = (off + bytes + 255) & ~(size_t)255;
    return p;
  };
  float*    p1   = (float*)   alloc((size_t)N_NODES*20*144*sizeof(float)); // 47.2 MB
  float*    p2   = (float*)   alloc((size_t)N_NODES*800*sizeof(float));    // 13.1 MB
  float*    h    = (float*)   alloc((size_t)N_NODES*FEAT*sizeof(float));
  _Float16* hpad = (_Float16*)alloc((size_t)N_NODES*FPAD*sizeof(_Float16));
  float*    sq   = (float*)   alloc((size_t)N_NODES*sizeof(float));
  float*    Z    = (float*)   alloc((size_t)N_NODES*FEAT*sizeof(float));
  float*    agg  = (float*)   alloc((size_t)N_NODES*FEAT*sizeof(float));
  float*    deg  = (float*)   alloc((size_t)N_NODES*sizeof(float));
  int*      hist = (int*)     alloc((NCAND+2)*sizeof(int));
  float*    sel  = (float*)   alloc(256);

  k_init <<<1, 256, 0, stream>>>(hist);
  k_conv1<<<(N_NODES*20*144 + 255)/256, 256, 0, stream>>>(x, c1w, c1b, p1);
  k_conv2<<<N_NODES, 256, 0, stream>>>(p1, c2w, c2b, p2);
  k_fc   <<<N_NODES, 64, 0, stream>>>(p2, fcw, fcb, h, hpad, sq);
  k_gram_hist<<<(NT*NT)/8, 256, 0, stream>>>(hpad, sq, hist);
  k_select<<<1, 1, 0, stream>>>(hist, sel);
  k_spmm <<<NT, 256, 0, stream>>>(hpad, sq, h, sel, agg, deg);
  k_sage <<<N_NODES, 64, 0, stream>>>(h, agg, deg, erel_w, erel_b, eroot_w, Z);
  k_final<<<1, 64, 0, stream>>>(Z, sel, erel_w, erel_b, eroot_w,
                                l1w, l1b, l2w, l2b, (float*)d_out, out_size);
}
